// TransformerBlock_1949915152691
// MI455X (gfx1250) — matrix operand order repaired and checked by reading, not compiled, not run on hardware
//
#include <hip/hip_runtime.h>
#include <math.h>

#define B_ 2
#define N_ 2048
#define D_ 1024
#define H_ 16
#define E_ 64
#define F_ 4096

typedef unsigned short u16;
typedef unsigned int   u32;
typedef __attribute__((ext_vector_type(16))) __bf16 v16bf;
typedef __attribute__((ext_vector_type(8)))  float  v8f;
typedef __attribute__((ext_vector_type(4)))  int    v4i;

union FragBF { uint4 q[2]; v16bf v; };

#if __has_builtin(__builtin_amdgcn_global_load_async_to_lds_b128)
#define USE_ASYNC_LDS 1
#else
#define USE_ASYNC_LDS 0
#endif

typedef __attribute__((address_space(1))) v4i* gptr_b128;
typedef __attribute__((address_space(3))) v4i* lptr_b128;

__device__ __forceinline__ u16 f2bf(float x) {
  union { float f; u32 u; } c; c.f = x;
  u32 u = c.u;
  return (u16)((u + 0x7FFFu + ((u >> 16) & 1u)) >> 16);  // RNE
}

__device__ __forceinline__ v8f wmma_bf16(v16bf a, v16bf b, v8f c) {
  return __builtin_amdgcn_wmma_f32_16x16x32_bf16(false, a, false, b,
                                                 (short)0, c, false, false);
}

__device__ __forceinline__ void async_cp128(const u16* g, u16* l) {
#if USE_ASYNC_LDS
  __builtin_amdgcn_global_load_async_to_lds_b128(
      (gptr_b128)(u16*)g, (lptr_b128)l, 0, 0);
#else
  (void)g; (void)l;
#endif
}

__device__ __forceinline__ void wait_async0() {
#if __has_builtin(__builtin_amdgcn_s_wait_asynccnt)
  __builtin_amdgcn_s_wait_asynccnt(0);
#else
  asm volatile("s_wait_asynccnt 0" ::: "memory");
#endif
}

// ---------------- fp32 -> bf16 weight conversion ----------------
__global__ __launch_bounds__(256) void cvt_bf16_kernel(
    const float* __restrict__ in, u16* __restrict__ out, int n) {
  int i = blockIdx.x * 256 + threadIdx.x;
  if (i < n) out[i] = f2bf(in[i]);
}

// ---------------- block-wide sum (256 threads = 8 waves) ----------------
__device__ __forceinline__ float block_sum(float v, float* sh) {
  for (int off = 16; off > 0; off >>= 1) v += __shfl_xor(v, off, 32);
  int w = threadIdx.x >> 5, l = threadIdx.x & 31;
  __syncthreads();
  if (l == 0) sh[w] = v;
  __syncthreads();
  float tot = 0.f;
  for (int i = 0; i < 8; i++) tot += sh[i];
  return tot;
}

// ---------------- LayerNorm (optional residual add, optional ReLU) -> bf16 ----------------
template <int W, int PT, bool RELU>
__global__ __launch_bounds__(256) void ln_bf16_kernel(
    const float* __restrict__ x, const float* __restrict__ resid,
    const float* __restrict__ g, const float* __restrict__ bb,
    u16* __restrict__ out) {
  __shared__ float sh[8];
  const int row = blockIdx.x;
  const float* xr = x + (size_t)row * W;
  const float* rr = resid ? resid + (size_t)row * W : nullptr;
  float vals[PT];
  float s = 0.f;
  for (int i = 0; i < PT; i++) {
    int idx = threadIdx.x + i * 256;
    float v = xr[idx];
    if (rr) v += rr[idx];
    vals[i] = v; s += v;
  }
  float mu = block_sum(s, sh) * (1.0f / W);
  float s2 = 0.f;
  for (int i = 0; i < PT; i++) { float d = vals[i] - mu; s2 += d * d; }
  float var = block_sum(s2, sh) * (1.0f / W);
  float rstd = rsqrtf(var + 1e-5f);
  u16* orow = out + (size_t)row * W;
  for (int i = 0; i < PT; i++) {
    int idx = threadIdx.x + i * 256;
    float y = (vals[i] - mu) * rstd * g[idx] + bb[idx];
    if (RELU) y = fmaxf(y, 0.f);
    orow[idx] = f2bf(y);
  }
}

// ---------------- bf16 GEMM: C[M,Nout] = A[M,K] @ W[Nout,K]^T + bias ----------------
// 256 threads = 8 waves (2x4), WG tile 128x128, wave tile 64x32.
// K-step 64, ping-pong LDS, async global->LDS copies (ASYNCcnt) when available.
__global__ __launch_bounds__(256) void gemm_bf16_wmma(
    const u16* __restrict__ A, const u16* __restrict__ Wt,
    const float* __restrict__ bias,
    float* __restrict__ outF, u16* __restrict__ outB,
    int M, int K, int Nout) {
  __shared__ u16 As[2][128 * 72];  // 72-half stride: conflict-free b128 reads
  __shared__ u16 Ws[2][128 * 72];
  const int m0 = blockIdx.x * 128;
  const int n0 = blockIdx.y * 128;
  const int t = threadIdx.x;
  const int lane = t & 31;
  const int wid = t >> 5;
  const int wm = wid >> 2, wn = wid & 3;
  const int lh = lane & 15, hb = lane >> 4;
  const int kbA = hb * 8;    // A-frag K chunk base (ISA A 16x32 layout)
  const int kbB = hb * 8;    // B-frag K chunk base (same K order as A)
  const int crow = t & 127;          // copy: row
  const int ccol = (t >> 7) * 32;    // copy: 32-half chunk within row

  v8f acc[4][2];
  for (int i = 0; i < 4; i++)
    for (int j = 0; j < 2; j++)
      for (int r = 0; r < 8; r++) acc[i][j][r] = 0.f;

  const u16* agBase = A  + (size_t)(m0 + crow) * K + ccol;
  const u16* wgBase = Wt + (size_t)(n0 + crow) * K + ccol;
  u16* alBase[2] = { &As[0][crow * 72 + ccol], &As[1][crow * 72 + ccol] };
  u16* wlBase[2] = { &Ws[0][crow * 72 + ccol], &Ws[1][crow * 72 + ccol] };

  const int S = K >> 6;  // # of 64-wide K tiles

#if USE_ASYNC_LDS
  auto issue_copy = [&](int s, int b) {
    const u16* ag = agBase + (size_t)s * 64;
    const u16* wg = wgBase + (size_t)s * 64;
    u16* al = alBase[b];
    u16* wl = wlBase[b];
#pragma unroll
    for (int c = 0; c < 4; c++) {
      async_cp128(ag + c * 8, al + c * 8);
      async_cp128(wg + c * 8, wl + c * 8);
    }
  };
#else
  uint4 ra[4], rw[4];
  auto load_regs = [&](int s) {
    const u16* ag = agBase + (size_t)s * 64;
    const u16* wg = wgBase + (size_t)s * 64;
#pragma unroll
    for (int c = 0; c < 4; c++) {
      ra[c] = *(const uint4*)(ag + c * 8);
      rw[c] = *(const uint4*)(wg + c * 8);
    }
  };
  auto store_regs = [&](int b) {
    u16* al = alBase[b];
    u16* wl = wlBase[b];
#pragma unroll
    for (int c = 0; c < 4; c++) {
      *(uint4*)(al + c * 8) = ra[c];
      *(uint4*)(wl + c * 8) = rw[c];
    }
  };
#endif

  auto compute = [&](int b) {
#pragma unroll
    for (int kk = 0; kk < 2; kk++) {
      FragBF af[4], bfr[2];
#pragma unroll
      for (int mt = 0; mt < 4; mt++) {
        int r = wm * 64 + mt * 16 + lh;
        const u16* p = &As[b][r * 72 + kk * 32 + kbA];
        af[mt].q[0] = *(const uint4*)p;
        af[mt].q[1] = *(const uint4*)(p + 16);
      }
#pragma unroll
      for (int nt = 0; nt < 2; nt++) {
        int r = wn * 32 + nt * 16 + lh;
        const u16* p = &Ws[b][r * 72 + kk * 32 + kbB];
        bfr[nt].q[0] = *(const uint4*)p;
        bfr[nt].q[1] = *(const uint4*)(p + 16);
      }
#pragma unroll
      for (int mt = 0; mt < 4; mt++)
#pragma unroll
        for (int nt = 0; nt < 2; nt++)
          acc[mt][nt] = wmma_bf16(af[mt].v, bfr[nt].v, acc[mt][nt]);
    }
  };

#if USE_ASYNC_LDS
  issue_copy(0, 0);
  for (int s = 0; s < S; s++) {
    wait_async0();       // this wave's copies for tile s are resident
    __syncthreads();     // all waves' copies done + prior compute finished
    if (s + 1 < S) issue_copy(s + 1, (s + 1) & 1);  // overlaps compute(s)
    if (s + 2 < S) {
      __builtin_prefetch(agBase + (size_t)(s + 2) * 64, 0, 1);
      __builtin_prefetch(wgBase + (size_t)(s + 2) * 64, 0, 1);
    }
    compute(s & 1);
  }
#else
  load_regs(0);
  for (int s = 0; s < S; s++) {
    store_regs(s & 1);
    __syncthreads();
    if (s + 1 < S) load_regs(s + 1);  // overlaps compute(s)
    if (s + 2 < S) {
      __builtin_prefetch(agBase + (size_t)(s + 2) * 64, 0, 1);
      __builtin_prefetch(wgBase + (size_t)(s + 2) * 64, 0, 1);
    }
    compute(s & 1);
    __syncthreads();
  }
#endif

  for (int mt = 0; mt < 4; mt++) {
    for (int nt = 0; nt < 2; nt++) {
      int n = n0 + wn * 32 + nt * 16 + lh;
      float bv = bias ? bias[n] : 0.f;
      for (int r = 0; r < 8; r++) {
        int m = m0 + wm * 64 + mt * 16 + r + 8 * hb;  // C layout: rows r / r+8
        float val = acc[mt][nt][r] + bv;
        size_t off = (size_t)m * Nout + n;
        if (outF) outF[off] = val;
        if (outB) outB[off] = f2bf(val);
      }
    }
  }
}

// ---------------- v [B,N,H*E] -> vT [B,H,E,N] ----------------
__global__ __launch_bounds__(256) void vtrans_kernel(
    const u16* __restrict__ v, u16* __restrict__ vT) {
  int idx = blockIdx.x * 256 + threadIdx.x;
  if (idx >= B_ * N_ * D_) return;
  int d = idx & (D_ - 1);
  int n = (idx >> 10) & (N_ - 1);
  int b = idx >> 21;
  int h = d >> 6, e = d & 63;
  vT[((((size_t)b * H_ + h) * E_ + e) << 11) + n] = v[idx];
}

// ---------------- flash attention: one wave per 16-query tile ----------------
__global__ __launch_bounds__(128) void attn_fwd_kernel(
    const u16* __restrict__ qg, const u16* __restrict__ kg,
    const u16* __restrict__ vTg, float* __restrict__ attn) {
  __shared__ u16 plds[4][16 * 32];  // per-wave P re-striping scratch
  const int lwid = threadIdx.x >> 5;
  const int lane = threadIdx.x & 31;
  const int gwid = blockIdx.x * 4 + lwid;
  const int bh = gwid >> 7;             // / (N/16)
  const int q0 = (gwid & 127) * 16;
  const int b = bh >> 4, h = bh & 15;
  const int lh = lane & 15, hb = lane >> 4;
  u16* pl = plds[lwid];
  const size_t base = ((size_t)b * N_) * D_ + h * E_;
  const size_t vtb  = ((size_t)bh * E_) * N_;

  // Q fragments (A-layout), E split into two K=32 steps
  FragBF qa[2];
  {
    const u16* qp = qg + base + (size_t)(q0 + lh) * D_ + hb * 8;
    for (int es = 0; es < 2; es++) {
      qa[es].q[0] = *(const uint4*)(qp + es * 32);
      qa[es].q[1] = *(const uint4*)(qp + es * 32 + 16);
    }
  }
  float mrun[8], lrun[8], alpha[8];
  v8f o[4];
  for (int r = 0; r < 8; r++) { mrun[r] = -INFINITY; lrun[r] = 0.f; }
  for (int et = 0; et < 4; et++)
    for (int r = 0; r < 8; r++) o[et][r] = 0.f;

  const int kend = q0 + 16;  // causal: keys 0 .. q0+15
  for (int j0 = 0; j0 < kend; j0 += 32) {
    // S[16,32] = Q @ K^T : two 16x16 column tiles, E reduced in two wmmas
    v8f st[2];
    for (int nt = 0; nt < 2; nt++) {
      const u16* kp = kg + base + (size_t)(j0 + nt * 16 + lh) * D_ + hb * 8;
      FragBF k0f, k1f;
      k0f.q[0] = *(const uint4*)kp;
      k0f.q[1] = *(const uint4*)(kp + 16);
      k1f.q[0] = *(const uint4*)(kp + 32);
      k1f.q[1] = *(const uint4*)(kp + 48);
      v8f c; for (int i = 0; i < 8; i++) c[i] = 0.f;
      c = wmma_bf16(qa[0].v, k0f.v, c);
      c = wmma_bf16(qa[1].v, k1f.v, c);
      st[nt] = c;
    }
    // streaming causal softmax (row stats via 16-lane butterflies)
    for (int r = 0; r < 8; r++) {
      const int qrow = q0 + r + 8 * hb;
      float x0 = st[0][r] * 32.0f;               // * sqrt(D)
      float x1 = st[1][r] * 32.0f;
      if (j0 + lh > qrow)      x0 = -INFINITY;
      if (j0 + 16 + lh > qrow) x1 = -INFINITY;
      float rmx = fmaxf(x0, x1);
      for (int off = 1; off < 16; off <<= 1)
        rmx = fmaxf(rmx, __shfl_xor(rmx, off, 32));
      float mnew = fmaxf(mrun[r], rmx);
      float a = __expf(mrun[r] - mnew);
      float p0 = __expf(x0 - mnew);
      float p1 = __expf(x1 - mnew);
      float ps = p0 + p1;
      for (int off = 1; off < 16; off <<= 1) ps += __shfl_xor(ps, off, 32);
      lrun[r] = lrun[r] * a + ps;
      mrun[r] = mnew;
      alpha[r] = a;
      const int mrow = r + 8 * hb;
      pl[mrow * 32 + lh]      = f2bf(p0);
      pl[mrow * 32 + 16 + lh] = f2bf(p1);
    }
    // wave-internal LDS ordering: C-layout stores -> A-layout loads
    asm volatile("s_wait_dscnt 0" ::: "memory");
    FragBF pa;
    pa.q[0] = *(const uint4*)&pl[lh * 32 + hb * 8];
    pa.q[1] = *(const uint4*)&pl[lh * 32 + hb * 8 + 16];
    // O += P @ V  (V pre-transposed so B-frag loads are contiguous)
    for (int et = 0; et < 4; et++) {
      for (int r = 0; r < 8; r++) o[et][r] *= alpha[r];
      FragBF vf;
      const u16* vp = vTg + vtb + (size_t)(et * 16 + lh) * N_ + j0 + hb * 8;
      vf.q[0] = *(const uint4*)vp;
      vf.q[1] = *(const uint4*)(vp + 16);
      o[et] = wmma_bf16(pa.v, vf.v, o[et]);
    }
  }
  for (int et = 0; et < 4; et++) {
    for (int r = 0; r < 8; r++) {
      const int qrow = q0 + r + 8 * hb;
      attn[base + (size_t)qrow * D_ + et * 16 + lh] = o[et][r] / lrun[r];
    }
  }
}

// ---------------- host orchestration ----------------
extern "C" void kernel_launch(void* const* d_in, const int* in_sizes, int n_in,
                              void* d_out, int out_size, void* d_ws, size_t ws_size,
                              hipStream_t stream) {
  (void)in_sizes; (void)n_in; (void)out_size; (void)ws_size;
  const float* x    = (const float*)d_in[0];
  const float* wq   = (const float*)d_in[1];
  const float* bq   = (const float*)d_in[2];
  const float* wk   = (const float*)d_in[3];
  const float* bk   = (const float*)d_in[4];
  const float* wv   = (const float*)d_in[5];
  const float* bv   = (const float*)d_in[6];
  const float* w1   = (const float*)d_in[7];
  const float* b1   = (const float*)d_in[8];
  const float* lng  = (const float*)d_in[9];
  const float* lnb  = (const float*)d_in[10];
  const float* w2   = (const float*)d_in[11];
  const float* b2   = (const float*)d_in[12];
  const float* ln1g = (const float*)d_in[13];
  const float* ln1b = (const float*)d_in[14];
  const float* ln2g = (const float*)d_in[15];
  const float* ln2b = (const float*)d_in[16];

  const size_t ND = (size_t)B_ * N_ * D_;   // 4M elems
  char* ws = (char*)d_ws;
  u16* wq_bf = (u16*)ws; ws += (size_t)D_ * D_ * 2;
  u16* wk_bf = (u16*)ws; ws += (size_t)D_ * D_ * 2;
  u16* wv_bf = (u16*)ws; ws += (size_t)D_ * D_ * 2;
  u16* w1_bf = (u16*)ws; ws += (size_t)F_ * D_ * 2;
  u16* w2_bf = (u16*)ws; ws += (size_t)D_ * F_ * 2;
  u16* h_bf  = (u16*)ws; ws += ND * 2;      // reused later as h2
  char* big  = ws;                          // 64MB region, reused for MLP mid
  u16* q_bf  = (u16*)big;
  u16* k_bf  = q_bf + ND;
  u16* v_bf  = k_bf + ND;
  u16* vT_bf = v_bf + ND;
  float* attn = (float*)(vT_bf + ND);
  float* m_f  = (float*)big;                // overlaps q/k/v/vT/attn (dead by then)
  u16* m_bf   = (u16*)(big + (size_t)4096 * 4096 * 4);

  const int NT = 256;
  cvt_bf16_kernel<<<(D_ * D_ + NT - 1) / NT, NT, 0, stream>>>(wq, wq_bf, D_ * D_);
  cvt_bf16_kernel<<<(D_ * D_ + NT - 1) / NT, NT, 0, stream>>>(wk, wk_bf, D_ * D_);
  cvt_bf16_kernel<<<(D_ * D_ + NT - 1) / NT, NT, 0, stream>>>(wv, wv_bf, D_ * D_);
  cvt_bf16_kernel<<<(F_ * D_ + NT - 1) / NT, NT, 0, stream>>>(w1, w1_bf, F_ * D_);
  cvt_bf16_kernel<<<(D_ * F_ + NT - 1) / NT, NT, 0, stream>>>(w2, w2_bf, D_ * F_);

  ln_bf16_kernel<D_, 4, false><<<B_ * N_, 256, 0, stream>>>(x, nullptr, ln1g, ln1b, h_bf);

  dim3 gqkv(32, 8);
  gemm_bf16_wmma<<<gqkv, 256, 0, stream>>>(h_bf, wq_bf, bq, nullptr, q_bf, 4096, 1024, 1024);
  gemm_bf16_wmma<<<gqkv, 256, 0, stream>>>(h_bf, wk_bf, bk, nullptr, k_bf, 4096, 1024, 1024);
  gemm_bf16_wmma<<<gqkv, 256, 0, stream>>>(h_bf, wv_bf, bv, nullptr, v_bf, 4096, 1024, 1024);

  vtrans_kernel<<<(int)((ND + NT - 1) / NT), NT, 0, stream>>>(v_bf, vT_bf);

  attn_fwd_kernel<<<(B_ * H_ * (N_ / 16)) / 4, 128, 0, stream>>>(q_bf, k_bf, vT_bf, attn);

  ln_bf16_kernel<D_, 4, false><<<B_ * N_, 256, 0, stream>>>(x, attn, ln2g, ln2b, h_bf);

  gemm_bf16_wmma<<<dim3(32, 32), 256, 0, stream>>>(h_bf, w1_bf, b1, m_f, nullptr, 4096, 1024, 4096);

  ln_bf16_kernel<F_, 16, true><<<B_ * N_, 256, 0, stream>>>(m_f, nullptr, lng, lnb, m_bf);

  gemm_bf16_wmma<<<dim3(32, 8), 256, 0, stream>>>(m_bf, w2_bf, b2, (float*)d_out, nullptr, 4096, 4096, 1024);
}
